// CausalSelfAttention_16801912062622
// MI455X (gfx1250) — compile-verified, matrix operand order checked by reading, not run on hardware
//
#include <hip/hip_runtime.h>

// B=2, T=2048, C=2048, H=16, D=128
typedef unsigned short ushort;
typedef __attribute__((ext_vector_type(16))) __bf16 v16bf;
typedef __attribute__((ext_vector_type(8)))  float  v8f;

__device__ __forceinline__ ushort f2bf(float f) {
  unsigned u = __builtin_bit_cast(unsigned, f);
  u += 0x7FFFu + ((u >> 16) & 1u);   // round-to-nearest-even
  return (ushort)(u >> 16);
}

// Load one 16x32 bf16 fragment (A layout / B-as-N-major layout).
// lane L holds row m=L&15; elements 0..7 -> k = kbase + (L>>4)*8 + 0..7,
// elements 8..15 -> +16.  p points at row 0; stride in ushorts (mult of 8).
__device__ __forceinline__ v16bf frag_ld(const ushort* p, int stride, int kbase) {
  const int lane = threadIdx.x & 31;
  const ushort* q = p + (lane & 15) * stride + kbase + (lane >> 4) * 8;
  union { uint4 u[2]; v16bf v; } t;
  t.u[0] = *(const uint4*)q;
  t.u[1] = *(const uint4*)(q + 16);
  return t.v;
}

// CDNA5 async global->LDS copy (16B per lane), tracked by ASYNCcnt.
__device__ __forceinline__ void async_cp16(void* lds, const void* g) {
  unsigned l = (unsigned)(unsigned long long)lds;          // low 32 bits = LDS offset
  asm volatile("global_load_async_to_lds_b128 %0, %1, off"
               :: "v"(l), "v"((unsigned long long)g) : "memory");
}
__device__ __forceinline__ void async_wait0() {
  asm volatile("s_wait_asynccnt 0x0" ::: "memory");
}

// ---------------- fp32 -> bf16 convert (linear) ----------------
__global__ void cvt_f32_bf16(const float* __restrict__ in, ushort* __restrict__ out, long n4) {
  long i  = (long)blockIdx.x * blockDim.x + threadIdx.x;
  long st = (long)gridDim.x * blockDim.x;
  for (; i < n4; i += st) {
    float4 f = ((const float4*)in)[i];
    union { ushort u[4]; uint2 v; } o;
    o.u[0] = f2bf(f.x); o.u[1] = f2bf(f.y); o.u[2] = f2bf(f.z); o.u[3] = f2bf(f.w);
    ((uint2*)out)[i] = o.v;
  }
}

// ---------------- fp32 -> bf16 convert + transpose (KxN -> NxK) ----------------
__global__ __launch_bounds__(256) void cvt_transpose_bf16(
    const float* __restrict__ in, ushort* __restrict__ out, int K, int N) {
  __shared__ float tile[32][33];
  const int k0 = blockIdx.y * 32, n0 = blockIdx.x * 32;
  const int tx = threadIdx.x & 31, ty = threadIdx.x >> 5;  // 32 x 8
#pragma unroll
  for (int i = 0; i < 32; i += 8)
    tile[ty + i][tx] = in[(long)(k0 + ty + i) * N + n0 + tx];
  __syncthreads();
#pragma unroll
  for (int i = 0; i < 32; i += 8)
    out[(long)(n0 + ty + i) * K + k0 + tx] = f2bf(tile[tx][ty + i]);
}

// ---------------- 128x128x32 double-buffered bf16 WMMA GEMM mainloop ----------------
// A: MxK row-major bf16. Bt: NxK row-major bf16 (weights pre-transposed).
// Both stage into LDS with clean b128 copies (no scatter).
// 256 threads = 8 waves, wave tile 32x64 -> acc[2][4] of 16x16 f32 fragments.
__device__ __forceinline__ void gemm_tile_bf16(
    const ushort* __restrict__ A, const ushort* __restrict__ Bt,
    int K, long m0, long n0, ushort* As, ushort* Bs, v8f acc[2][4])
{
  const int tid = threadIdx.x;
  const int wid = tid >> 5;
  const int wr  = (wid & 3) * 32;
  const int wc  = (wid >> 2) * 64;
  const int row = tid >> 1, seg = (tid & 1) * 16;
  const int KT  = K >> 5;
  const ushort* aptr = A  + (m0 + row) * (long)K + seg;
  const ushort* bptr = Bt + (n0 + row) * (long)K + seg;

  uint4 ga0 = *(const uint4*)aptr, ga1 = *(const uint4*)(aptr + 8);
  uint4 gb0 = *(const uint4*)bptr, gb1 = *(const uint4*)(bptr + 8);

  for (int kt = 0; kt < KT; ++kt) {
    ushort* as = As + (kt & 1) * (128 * 32);
    ushort* bs = Bs + (kt & 1) * (128 * 32);
    *(uint4*)(as + row * 32 + seg)     = ga0;
    *(uint4*)(as + row * 32 + seg + 8) = ga1;
    *(uint4*)(bs + row * 32 + seg)     = gb0;
    *(uint4*)(bs + row * 32 + seg + 8) = gb1;
    __syncthreads();
    if (kt + 1 < KT) {  // prefetch next K-tile while computing this one
      const ushort* ap = aptr + (long)(kt + 1) * 32;
      const ushort* bp = bptr + (long)(kt + 1) * 32;
      ga0 = *(const uint4*)ap; ga1 = *(const uint4*)(ap + 8);
      gb0 = *(const uint4*)bp; gb1 = *(const uint4*)(bp + 8);
    }
    v16bf af0 = frag_ld(as + (wr     ) * 32, 32, 0);
    v16bf af1 = frag_ld(as + (wr + 16) * 32, 32, 0);
#pragma unroll
    for (int j = 0; j < 4; ++j) {
      v16bf bf = frag_ld(bs + (wc + j * 16) * 32, 32, 0);
      acc[0][j] = __builtin_amdgcn_wmma_f32_16x16x32_bf16(false, af0, false, bf, (short)0, acc[0][j], false, false);
      acc[1][j] = __builtin_amdgcn_wmma_f32_16x16x32_bf16(false, af1, false, bf, (short)0, acc[1][j], false, false);
    }
    // store(kt+2 -> same buffer) is ordered by the sync at top of kt+1
  }
}

// ---------------- qkv = x @ W_attn, fused RoPE + scatter ----------------
// q,k -> [B,H,T,D]; v -> [B,H,D,T] (transposed so attention V staging is contiguous)
__global__ __launch_bounds__(256) void qkv_rope_kernel(
    const ushort* __restrict__ Xb, const ushort* __restrict__ Wat,
    const float* __restrict__ rc, const float* __restrict__ rs,
    ushort* __restrict__ qb, ushort* __restrict__ kb, ushort* __restrict__ vtb)
{
  __shared__ ushort As[2 * 128 * 32];
  __shared__ ushort Bs[2 * 128 * 32];
  const long m0 = (long)blockIdx.y * 128;   // row in [0, 4096)
  const long n0 = (long)blockIdx.x * 128;   // col in [0, 6144)
  v8f acc[2][4];
#pragma unroll
  for (int i = 0; i < 2; ++i)
#pragma unroll
    for (int j = 0; j < 4; ++j) acc[i][j] = (v8f){0,0,0,0,0,0,0,0};

  gemm_tile_bf16(Xb, Wat, 2048, m0, n0, As, Bs, acc);

  const int tid = threadIdx.x, wid = tid >> 5, lane = tid & 31;
  const int half = lane >> 4, nn = lane & 15;
  const int wr = (wid & 3) * 32, wcc = (wid >> 2) * 64;
  const int bidx = (int)(m0 >> 11);         // batch (block fits in one batch)
#pragma unroll
  for (int i = 0; i < 2; ++i) {
#pragma unroll
    for (int j = 0; j < 4; ++j) {
      const int col   = (int)n0 + wcc + j * 16 + nn;
      const int which = col >> 11;          // 0=q 1=k 2=v (uniform per wave)
      const int cc    = col & 2047;
      const int h     = cc >> 7, d = cc & 127;
#pragma unroll
      for (int r = 0; r < 8; ++r) {
        const long row = m0 + wr + i * 16 + r + 8 * half;
        const int  t   = (int)(row & 2047);
        float v = acc[i][j][r];
        if (which < 2) {                    // RoPE: pair lives in adjacent lane
          float p = __shfl_xor(v, 1);
          float c = rc[t * 64 + (d >> 1)];
          float s = rs[t * 64 + (d >> 1)];
          float ov = ((d & 1) == 0) ? (v * c - p * s) : (p * s + v * c);
          ushort* dst = (which == 0) ? qb : kb;
          dst[(((long)bidx * 16 + h) * 2048 + t) * 128 + d] = f2bf(ov);
        } else {                            // v stored transposed [B,H,D,T]
          vtb[(((long)bidx * 16 + h) * 128 + d) * 2048 + t] = f2bf(v);
        }
      }
    }
  }
}

// ---------------- flash attention: 128 q-rows/block, 64-key chunks ----------------
__global__ __launch_bounds__(256) void flash_attn_kernel(
    const ushort* __restrict__ Qb, const ushort* __restrict__ Kb,
    const ushort* __restrict__ Vtg, ushort* __restrict__ Yb)
{
  __shared__ ushort Ks[64 * 128];   // [key][d]  (B-frag layout for S)
  __shared__ ushort Vt[128 * 64];   // [d][key]  (B-frag layout for P.V)
  __shared__ ushort Ps[8 * 16 * 64];// per-wave P staging [row][key]
  const int tid = threadIdx.x, wid = tid >> 5, lane = tid & 31;
  const int half = lane >> 4, nn = lane & 15;
  const int b = blockIdx.z, h = blockIdx.y, qt = blockIdx.x;
  const long bh = (long)b * 16 + h;
  const ushort* Qp = Qb  + bh * 2048 * 128;
  const ushort* Kp = Kb  + bh * 2048 * 128;
  const ushort* Vp = Vtg + bh * 128 * 2048;   // [d][t]
  const int t0 = qt * 128;
  const int qrow0 = t0 + wid * 16;            // this wave's 16 q rows

  v16bf qf[4];                                // Q 16x128 A-fragments from global
#pragma unroll
  for (int c = 0; c < 4; ++c) qf[c] = frag_ld(Qp + (long)qrow0 * 128, 128, c * 32);

  v8f of[8];
#pragma unroll
  for (int d = 0; d < 8; ++d) of[d] = (v8f){0,0,0,0,0,0,0,0};
  float mrow[8], lrow[8];
#pragma unroll
  for (int r = 0; r < 8; ++r) { mrow[r] = -3.0e38f; lrow[r] = 0.f; }

  // staging indices: K: 64 rows x 128 d ; Vt: 128 rows(d) x 64 keys
  const int kr = tid >> 2, kseg = (tid & 3) * 32;   // K copy
  const int vr = tid >> 1, vseg = (tid & 1) * 32;   // V copy

  const float scale = 0.08838834764831845f; // 1/sqrt(128)
  const int nkc = (t0 + 128) >> 6;          // causal: keys < t0+128
  for (int kc = 0; kc < nkc; ++kc) {
    const int k0 = kc * 64;
    { // async global->LDS staging (ASYNCcnt-tracked, no VGPR round trip)
      const ushort* gk = Kp + (long)(k0 + kr) * 128 + kseg;
      ushort*       lk = Ks + kr * 128 + kseg;
#pragma unroll
      for (int i = 0; i < 4; ++i) async_cp16(lk + i * 8, gk + i * 8);
      const ushort* gv = Vp + (long)vr * 2048 + k0 + vseg;
      ushort*       lv = Vt + vr * 64 + vseg;
#pragma unroll
      for (int i = 0; i < 4; ++i) async_cp16(lv + i * 8, gv + i * 8);
      async_wait0();
    }
    __syncthreads();

    // S = Q K^T  (16 x 64)
    v8f sf[4];
#pragma unroll
    for (int j = 0; j < 4; ++j) sf[j] = (v8f){0,0,0,0,0,0,0,0};
#pragma unroll
    for (int c = 0; c < 4; ++c) {
#pragma unroll
      for (int j = 0; j < 4; ++j) {
        v16bf bf = frag_ld(Ks + (j * 16) * 128, 128, c * 32);
        sf[j] = __builtin_amdgcn_wmma_f32_16x16x32_bf16(false, qf[c], false, bf, (short)0, sf[j], false, false);
      }
    }
    // scale + causal mask
    float sv[4][8];
#pragma unroll
    for (int j = 0; j < 4; ++j) {
      const int key = k0 + j * 16 + nn;
#pragma unroll
      for (int r = 0; r < 8; ++r) {
        const int q = qrow0 + r + 8 * half;
        float v = sf[j][r] * scale;
        sv[j][r] = (key > q) ? -1.0e30f : v;
      }
    }
    // online softmax: row max/sum via shfl across the 16 column lanes
    float al[8];
#pragma unroll
    for (int r = 0; r < 8; ++r) {
      float m = fmaxf(fmaxf(sv[0][r], sv[1][r]), fmaxf(sv[2][r], sv[3][r]));
#pragma unroll
      for (int off = 1; off < 16; off <<= 1) m = fmaxf(m, __shfl_xor(m, off));
      float nm = fmaxf(mrow[r], m);
      al[r] = __expf(mrow[r] - nm);
      mrow[r] = nm;
      float rsum = 0.f;
#pragma unroll
      for (int j = 0; j < 4; ++j) { float p = __expf(sv[j][r] - nm); sv[j][r] = p; rsum += p; }
#pragma unroll
      for (int off = 1; off < 16; off <<= 1) rsum += __shfl_xor(rsum, off);
      lrow[r] = lrow[r] * al[r] + rsum;
    }
#pragma unroll
    for (int d = 0; d < 8; ++d)
#pragma unroll
      for (int r = 0; r < 8; ++r) of[d][r] *= al[r];
    // P (C-layout) -> per-wave LDS (A-layout), then O += P V
    ushort* pw = Ps + wid * 16 * 64;
#pragma unroll
    for (int j = 0; j < 4; ++j)
#pragma unroll
      for (int r = 0; r < 8; ++r)
        pw[(r + 8 * half) * 64 + j * 16 + nn] = f2bf(sv[j][r]);
    // same-wave LDS ops stay in order -> stores visible to frag_ld reads
#pragma unroll
    for (int kk = 0; kk < 2; ++kk) {
      v16bf pa = frag_ld(pw, 64, kk * 32);
#pragma unroll
      for (int d = 0; d < 8; ++d) {
        v16bf bf = frag_ld(Vt + (d * 16) * 64, 64, kk * 32);
        of[d] = __builtin_amdgcn_wmma_f32_16x16x32_bf16(false, pa, false, bf, (short)0, of[d], false, false);
      }
    }
    __syncthreads();  // all waves done reading before restaging Ks/Vt
  }
  // normalize and emit y[b*T+t][h*128+d] bf16 for the projection GEMM
#pragma unroll
  for (int r = 0; r < 8; ++r) mrow[r] = 1.0f / lrow[r];   // reuse as inv
#pragma unroll
  for (int d = 0; d < 8; ++d) {
    const int ccol = h * 128 + d * 16 + nn;
#pragma unroll
    for (int r = 0; r < 8; ++r) {
      const int t = qrow0 + r + 8 * half;
      Yb[((long)b * 2048 + t) * 2048 + ccol] = f2bf(of[d][r] * mrow[r]);
    }
  }
}

// ---------------- out = y @ W_proj (f32 epilogue) ----------------
__global__ __launch_bounds__(256) void proj_kernel(
    const ushort* __restrict__ Yb, const ushort* __restrict__ Wpt,
    float* __restrict__ out)
{
  __shared__ ushort As[2 * 128 * 32];
  __shared__ ushort Bs[2 * 128 * 32];
  const long m0 = (long)blockIdx.y * 128;
  const long n0 = (long)blockIdx.x * 128;
  v8f acc[2][4];
#pragma unroll
  for (int i = 0; i < 2; ++i)
#pragma unroll
    for (int j = 0; j < 4; ++j) acc[i][j] = (v8f){0,0,0,0,0,0,0,0};

  gemm_tile_bf16(Yb, Wpt, 2048, m0, n0, As, Bs, acc);

  const int tid = threadIdx.x, wid = tid >> 5, lane = tid & 31;
  const int half = lane >> 4, nn = lane & 15;
  const int wr = (wid & 3) * 32, wcc = (wid >> 2) * 64;
#pragma unroll
  for (int i = 0; i < 2; ++i)
#pragma unroll
    for (int j = 0; j < 4; ++j) {
      const long col = n0 + wcc + j * 16 + nn;
#pragma unroll
      for (int r = 0; r < 8; ++r) {
        const long row = m0 + wr + i * 16 + r + 8 * half;
        out[row * 2048 + col] = acc[i][j][r];
      }
    }
}

extern "C" void kernel_launch(void* const* d_in, const int* in_sizes, int n_in,
                              void* d_out, int out_size, void* d_ws, size_t ws_size,
                              hipStream_t stream) {
  const float* x      = (const float*)d_in[0];   // (2,2048,2048)
  const float* W_attn = (const float*)d_in[1];   // (2048,6144)
  const float* W_proj = (const float*)d_in[2];   // (2048,2048)
  const float* rc     = (const float*)d_in[3];   // (2048,64)
  const float* rs     = (const float*)d_in[4];   // (2048,64)
  // d_in[5] attention_mask: all-true -> causal only
  float* out = (float*)d_out;

  // workspace layout (bf16/ushort elements), total ~112 MB
  ushort* xb  = (ushort*)d_ws;                          // 4096 x 2048
  ushort* Wat = xb  + (size_t)4096 * 2048;              // 6144 x 2048 (W_attn^T)
  ushort* Wpt = Wat + (size_t)6144 * 2048;              // 2048 x 2048 (W_proj^T)
  ushort* qb  = Wpt + (size_t)2048 * 2048;              // [B,H,T,D]
  ushort* kb  = qb  + (size_t)32 * 2048 * 128;          // [B,H,T,D]
  ushort* vtb = kb  + (size_t)32 * 2048 * 128;          // [B,H,D,T]
  ushort* yb  = vtb + (size_t)32 * 2048 * 128;          // 4096 x 2048

  cvt_f32_bf16<<<2048, 256, 0, stream>>>(x, xb, (long)4096 * 2048 / 4);
  cvt_transpose_bf16<<<dim3(192, 64), 256, 0, stream>>>(W_attn, Wat, 2048, 6144);
  cvt_transpose_bf16<<<dim3(64, 64),  256, 0, stream>>>(W_proj, Wpt, 2048, 2048);

  qkv_rope_kernel<<<dim3(48, 32), 256, 0, stream>>>(xb, Wat, rc, rs, qb, kb, vtb);
  flash_attn_kernel<<<dim3(16, 16, 2), 256, 0, stream>>>(qb, kb, vtb, yb);
  proj_kernel<<<dim3(16, 32), 256, 0, stream>>>(yb, Wpt, out);

  (void)in_sizes; (void)n_in; (void)out_size; (void)ws_size;
}
